// READOUTWEIGHTSPEmbedder3Conv_21062519620292
// MI455X (gfx1250) — compile-verified, matrix operand order checked by reading, not run on hardware
//
#include <hip/hip_runtime.h>

#define WG     256
#define NWAVE  8
#define NB     2048
#define NNODE  50
#define NEDGE  500
#define FIN    64
#define HID    256
#define NPAD   64

typedef __attribute__((ext_vector_type(16))) __bf16 v16bf;
typedef __attribute__((ext_vector_type(8)))  __bf16 v8bf;
typedef __attribute__((ext_vector_type(8)))  float  v8f;

__device__ __forceinline__ unsigned short f2bf(float f) {
  unsigned u = __float_as_uint(f);
  u += 0x7FFFu + ((u >> 16) & 1u);          // round-to-nearest-even
  return (unsigned short)(u >> 16);
}
__device__ __forceinline__ float bf2f(unsigned short h) {
  return __uint_as_float(((unsigned)h) << 16);
}
__device__ __forceinline__ float lrelu(float v) { return v > 0.f ? v : 0.01f * v; }

// Load one 16x32 bf16 WMMA fragment (A or B role) per ISA 7.12.2 layout:
// lanes 0-15: row = r0+lane,   elems 0..7 = K k0..k0+7,   elems 8..15 = K k0+16..k0+23
// lanes16-31: row = r0+lane-16,elems 0..7 = K k0+8..k0+15,elems 8..15 = K k0+24..k0+31
__device__ __forceinline__ v16bf load_frag(const unsigned short* base, int r0, int ld,
                                           int k0, int lane) {
  const int r  = r0 + (lane & 15);
  const int kk = k0 + ((lane >> 4) << 3);
  const unsigned short* p = base + r * ld + kk;
  union { v16bf v; v8bf h[2]; } u;
  u.h[0] = *(const v8bf*)(p);
  u.h[1] = *(const v8bf*)(p + 16);
  return u.v;
}

// MODE 0: out_bf16 = (acc * rowscale[row])            (conv message, pre-scatter)
// MODE 1: out_bf16 = relu(acc + bias[col])            (readout hidden)
// MODE 2: out_f32  = acc + bias[col]                  (readout final)
template<int KSTEPS, int MODE>
__device__ __forceinline__ void run_gemm(
    const unsigned short* A, int lda,          // LDS bf16, [64][lda]
    const unsigned short* Bm, int NT,          // global bf16, [NT*16][KSTEPS*32]
    float* outf, unsigned short* outb, int ldo,
    const float* rowscale, const float* bias,
    int wave, int lane)
{
  const int ldb = KSTEPS * 32;
  for (int nt = wave; nt < NT; nt += NWAVE) {
    const int n0 = nt * 16;
    v16bf bfrag[KSTEPS];
#pragma unroll
    for (int ks = 0; ks < KSTEPS; ++ks)
      bfrag[ks] = load_frag(Bm, n0, ldb, ks * 32, lane);
#pragma unroll
    for (int mt = 0; mt < 4; ++mt) {
      const int m0 = mt * 16;
      v8f acc = {0.f, 0.f, 0.f, 0.f, 0.f, 0.f, 0.f, 0.f};
#pragma unroll
      for (int ks = 0; ks < KSTEPS; ++ks) {
        v16bf afrag = load_frag(A, m0, lda, ks * 32, lane);
        acc = __builtin_amdgcn_wmma_f32_16x16x32_bf16(
            false, afrag, false, bfrag[ks], (short)0, acc, false, false);
      }
      // C/D layout: vgpr i -> row m0 + i + 8*(lane>=16), col n0 + (lane&15)
      const int col   = n0 + (lane & 15);
      const int rbase = m0 + ((lane >> 4) << 3);
#pragma unroll
      for (int i = 0; i < 8; ++i) {
        const int row = rbase + i;
        const float v = acc[i];
        if (MODE == 0)      outb[row * ldo + col] = f2bf(v * rowscale[row]);
        else if (MODE == 1) outb[row * ldo + col] = f2bf(fmaxf(v + bias[col], 0.f));
        else                outf[row * ldo + col] = v + bias[col];
      }
    }
  }
}

struct Params {
  const float* x;   const float* w;
  const int*   src; const int*   dst;
  const float* ar[4];
  const float* gamma[3]; const float* beta[3]; const float* alpha[3];
  const float* b1[3];    const float* b2[3];
  const unsigned short* Wt[3];     // [256][K]   bf16 (transposed conv weights)
  const unsigned short* rW1t[3];   // [128][256] bf16
  const unsigned short* rW2t[3];   // [64][128]  bf16
  float* out;
};

__global__ void convert_transpose_kernel(const float* __restrict__ in,
                                         unsigned short* __restrict__ out,
                                         int R, int C) {
  int i = blockIdx.x * blockDim.x + threadIdx.x;
  if (i >= R * C) return;
  int r = i / C, c = i - r * C;
  out[c * R + r] = f2bf(in[i]);
}

__global__ void __launch_bounds__(WG)
gnn_graph_kernel(Params P) {
  extern __shared__ char smem[];
  unsigned short* h_bf = (unsigned short*)smem;          // 64*256 bf16  (32 KB)
  unsigned short* msg  = h_bf + NPAD * HID;              // 64*256 bf16  (32 KB)
  float* agg           = (float*)(msg + NPAD * HID);     // 64*256 f32   (64 KB)
  int*   srcl          = (int*)(agg + NPAD * HID);       // 512 ints
  int*   dstl          = srcl + 512;                     // 512 ints
  float* oscale        = (float*)(dstl + 512);           // 64
  float* iscale        = oscale + NPAD;                  // 64
  float* warr          = iscale + NPAD;                  // 64

  const int b    = blockIdx.x;
  const int tid  = threadIdx.x;
  const int wave = tid >> 5;
  const int lane = tid & 31;

  // ---- stage edges, weights; zero LDS state ----
  for (int i = tid; i < NEDGE; i += WG) {
    srcl[i] = P.src[b * NEDGE + i];
    dstl[i] = P.dst[b * NEDGE + i];
  }
  if (tid < NPAD) {
    oscale[tid] = 0.f; iscale[tid] = 0.f;
    warr[tid] = (tid < NNODE) ? P.w[b * NNODE + tid] : 0.f;
  }
  for (int i = tid; i < NPAD * HID; i += WG) h_bf[i] = 0;
  __syncthreads();

  // ---- degrees via LDS float atomics ----
  for (int i = tid; i < NEDGE; i += WG) {
    atomicAdd(&oscale[srcl[i]], 1.f);
    atomicAdd(&iscale[dstl[i]], 1.f);
  }
  __syncthreads();
  if (tid < NPAD) {
    oscale[tid] = rsqrtf(fmaxf(oscale[tid], 1.f));
    iscale[tid] = rsqrtf(fmaxf(iscale[tid], 1.f));
  }
  // layer-0 activations: x -> bf16, ld = FIN
  for (int i = tid; i < NNODE * FIN; i += WG)
    h_bf[i] = f2bf(P.x[(size_t)b * NNODE * FIN + i]);
  __syncthreads();

  // ---- part 0: weighted mean of raw node feats (offset 0, 64 ch) ----
  if (tid < FIN) {
    float acc = 0.f;
    for (int n = 0; n < NNODE; ++n)
      acc += warr[n] * P.ar[0][n] * P.x[(size_t)b * NNODE * FIN + n * FIN + tid];
    P.out[(size_t)b * 1024 + tid] = lrelu(acc * (1.f / NNODE));
  }
  __syncthreads();

  for (int l = 0; l < 3; ++l) {
    // ---- GraphConv GEMM: msg = (h @ W) * outdeg^-1/2 ----
    if (l == 0)
      run_gemm<2, 0>(h_bf, FIN, P.Wt[0], 16, nullptr, msg, HID, oscale, nullptr, wave, lane);
    else
      run_gemm<8, 0>(h_bf, HID, P.Wt[l], 16, nullptr, msg, HID, oscale, nullptr, wave, lane);
    __syncthreads();

    // ---- scatter-add over edges (thread owns channel tid) ----
    for (int i = tid; i < NPAD * HID; i += WG) agg[i] = 0.f;
    __syncthreads();
    {
      const int c = tid;
      for (int e = 0; e < NEDGE; ++e) {
        const int s = srcl[e], d = dstl[e];
        agg[d * HID + c] += bf2f(msg[s * HID + c]);
      }
    }
    __syncthreads();

    // ---- indeg scale + GraphNorm + LeakyReLU + weighted-mean part ----
    {
      const int c = tid;
      const float ga = P.gamma[l][c], be = P.beta[l][c], al = P.alpha[l][c];
      const float* arv = P.ar[l + 1];
      float m = 0.f;
      for (int n = 0; n < NNODE; ++n) m += agg[n * HID + c] * iscale[n];
      m *= (1.f / NNODE);
      float var = 0.f;
      for (int n = 0; n < NNODE; ++n) {
        const float dd = agg[n * HID + c] * iscale[n] - al * m;
        var += dd * dd;
      }
      var *= (1.f / NNODE);
      const float rs = rsqrtf(var + 1e-5f);
      float wm = 0.f;
      for (int n = 0; n < NNODE; ++n) {
        float y = ga * (agg[n * HID + c] * iscale[n] - al * m) * rs + be;
        y = lrelu(y);
        h_bf[n * HID + c] = f2bf(y);
        wm += warr[n] * arv[n] * y;
      }
      P.out[(size_t)b * 1024 + 128 + l * 320 + c] = lrelu(wm * (1.f / NNODE));
    }
    __syncthreads();

    // ---- universal readout MLP (H -> H/2 -> H/4), mean over nodes ----
    unsigned short* u_bf = (unsigned short*)agg;           // 64*128 bf16 (16 KB)
    float* v_f = (float*)((char*)agg + 32768);             // 64*64  f32  (16 KB)
    run_gemm<8, 1>(h_bf, HID, P.rW1t[l], 8, nullptr, u_bf, 128, nullptr, P.b1[l], wave, lane);
    __syncthreads();
    run_gemm<4, 2>(u_bf, 128, P.rW2t[l], 4, v_f, nullptr, 64, nullptr, P.b2[l], wave, lane);
    __syncthreads();
    if (tid < 64) {
      float acc = 0.f;
      for (int n = 0; n < NNODE; ++n) acc += v_f[n * 64 + tid];
      P.out[(size_t)b * 1024 + 64 + l * 320 + tid] = lrelu(acc * (1.f / NNODE));
    }
    __syncthreads();
  }
}

extern "C" void kernel_launch(void* const* d_in, const int* in_sizes, int n_in,
                              void* d_out, int out_size, void* d_ws, size_t ws_size,
                              hipStream_t stream) {
  (void)in_sizes; (void)n_in; (void)out_size; (void)ws_size;

  // ---- bf16 transposed weights in workspace ----
  unsigned short* ws    = (unsigned short*)d_ws;
  unsigned short* W1t   = ws;               // [256][64]   16384
  unsigned short* W2t   = W1t + 16384;      // [256][256]  65536
  unsigned short* W3t   = W2t + 65536;      // [256][256]  65536
  unsigned short* r1W1t = W3t + 65536;      // [128][256]  32768
  unsigned short* r1W2t = r1W1t + 32768;    // [64][128]   8192
  unsigned short* r2W1t = r1W2t + 8192;
  unsigned short* r2W2t = r2W1t + 32768;
  unsigned short* r3W1t = r2W2t + 8192;
  unsigned short* r3W2t = r3W1t + 32768;

  auto cvt = [&](int idx, unsigned short* outp, int R, int C) {
    int n = R * C;
    convert_transpose_kernel<<<(n + 255) / 256, 256, 0, stream>>>(
        (const float*)d_in[idx], outp, R, C);
  };
  cvt(4,  W1t,   64,  256);   // W1  [F,H]  -> [H][F]
  cvt(5,  W2t,   256, 256);
  cvt(6,  W3t,   256, 256);
  cvt(14, r1W1t, 256, 128);   // r_W1 [H,H/2] -> [H/2][H]
  cvt(16, r1W2t, 128, 64);    // r_W2 [H/2,H/4] -> [H/4][H/2]
  cvt(21, r2W1t, 256, 128);
  cvt(23, r2W2t, 128, 64);
  cvt(28, r3W1t, 256, 128);
  cvt(30, r3W2t, 128, 64);

  Params P;
  P.x   = (const float*)d_in[0];
  P.w   = (const float*)d_in[1];
  P.src = (const int*)d_in[2];
  P.dst = (const int*)d_in[3];
  P.ar[0] = (const float*)d_in[7];
  P.ar[1] = (const float*)d_in[8];
  P.ar[2] = (const float*)d_in[9];
  P.ar[3] = (const float*)d_in[10];
  const unsigned short* W1s[3]  = {W1t, W2t, W3t};
  const unsigned short* rA[3]   = {r1W1t, r2W1t, r3W1t};
  const unsigned short* rB[3]   = {r1W2t, r2W2t, r3W2t};
  for (int l = 0; l < 3; ++l) {
    P.gamma[l] = (const float*)d_in[11 + 7 * l];
    P.beta[l]  = (const float*)d_in[12 + 7 * l];
    P.alpha[l] = (const float*)d_in[13 + 7 * l];
    P.b1[l]    = (const float*)d_in[15 + 7 * l];
    P.b2[l]    = (const float*)d_in[17 + 7 * l];
    P.Wt[l]    = W1s[l];
    P.rW1t[l]  = rA[l];
    P.rW2t[l]  = rB[l];
  }
  P.out = (float*)d_out;

  // 32KB h_bf16 + 32KB msg_bf16 + 64KB agg_f32 + edges/scales = 135936 B LDS
  const size_t smem = (size_t)NPAD * HID * 2 * 2 + (size_t)NPAD * HID * 4
                    + 1024 * 4 + 3 * NPAD * 4;
  gnn_graph_kernel<<<NB, WG, smem, stream>>>(P);
}
